// SimpleNeuralMemory_81458349736126
// MI455X (gfx1250) — matrix operand order repaired and checked by reading, not compiled, not run on hardware
//
#include <hip/hip_runtime.h>
#include <stdint.h>

// SimpleNeuralMemory for MI455X (gfx1250, wave32).
// Phase 1: q/k/v projections via bf16x3-split V_WMMA_F32_16X16X32_BF16 (f32-grade accuracy).
// Phase 2: column-parallel scan, state fully in VGPRs, one barrier/step, double-buffered LDS staging.

#define B_DIM 4
#define S_DIM 2048
#define D_DIM 768
#define NTOK  (B_DIM * S_DIM)   // 8192

typedef __bf16 v16bf __attribute__((ext_vector_type(16)));
typedef __bf16 v8bf  __attribute__((ext_vector_type(8)));
typedef float  v8f   __attribute__((ext_vector_type(8)));

__device__ __forceinline__ unsigned short f2bf_rne(float f) {
    unsigned u = __float_as_uint(f);
    unsigned r = u + 0x7FFFu + ((u >> 16) & 1u);   // round-to-nearest-even
    return (unsigned short)(r >> 16);
}
__device__ __forceinline__ float bf2f(unsigned short h) {
    return __uint_as_float(((unsigned)h) << 16);
}

// ---------------------------------------------------------------- split f32 -> (hi, lo) bf16
__global__ void split_bf16_kernel(const float* __restrict__ in,
                                  unsigned short* __restrict__ hi,
                                  unsigned short* __restrict__ lo, int n) {
    int i = blockIdx.x * blockDim.x + threadIdx.x;
    if (i < n) {
        float f = in[i];
        unsigned short h = f2bf_rne(f);
        hi[i] = h;
        lo[i] = f2bf_rne(f - bf2f(h));
    }
}

// ---------------------------------------------------------------- gates: sigmoid(x @ Wg.T + bg), f32
__global__ void gates_kernel(const float* __restrict__ x,
                             const float* __restrict__ Wg,
                             const float* __restrict__ bg,
                             float* __restrict__ gates) {
    int i = blockIdx.x * blockDim.x + threadIdx.x;      // over NTOK*3
    if (i >= NTOK * 3) return;
    int tok = i / 3, g = i % 3;
    const float* xr = x + (size_t)tok * D_DIM;
    const float* wr = Wg + (size_t)g * D_DIM;
    float acc = bg[g];
    for (int d = 0; d < D_DIM; ++d) acc = fmaf(xr[d], wr[d], acc);
    gates[i] = 1.0f / (1.0f + __expf(-acc));
}

// ---------------------------------------------------------------- projection GEMM: out = x @ W.T
// bf16x3: (xh+xl) x (wh+wl) ~= xh*wl + xl*wh + xh*wh, accumulated in f32 via WMMA.
// One wave computes one 16x16 output tile; 8 waves per block.
__global__ void __launch_bounds__(256)
proj_gemm_kernel(const unsigned short* __restrict__ xh,
                 const unsigned short* __restrict__ xl,
                 const unsigned short* __restrict__ wh,
                 const unsigned short* __restrict__ wl,
                 float* __restrict__ out) {
    const int lane = threadIdx.x & 31;
    const int wave = threadIdx.x >> 5;
    const int tile = blockIdx.x * 8 + wave;             // 24576 tiles total
    const int ntn  = D_DIM / 16;                        // 48 n-tiles
    const int m0 = (tile / ntn) * 16;
    const int n0 = (tile % ntn) * 16;
    const int hl = lane >> 4;                           // half-wave select
    const size_t arow = (size_t)(m0 + (lane & 15)) * D_DIM;  // A: lane holds token row
    const size_t brow = (size_t)(n0 + (lane & 15)) * D_DIM;  // B: lane holds W row (out feature)

    union U16 { v16bf v; v8bf h[2]; };
    v8f c = {};
    for (int k0 = 0; k0 < D_DIM; k0 += 32) {
        // A 16x32 bf16 layout: e0..15 <- K = k0 + 16*hl + 0..15
        const int ka = k0 + hl * 16;
        U16 ah, al;
        ah.h[0] = *(const v8bf*)(xh + arow + ka);
        ah.h[1] = *(const v8bf*)(xh + arow + ka + 8);
        al.h[0] = *(const v8bf*)(xl + arow + ka);
        al.h[1] = *(const v8bf*)(xl + arow + ka + 8);
        // B 32x16 bf16 layout: lanes 0-15 hold K = k0+0..15, lanes 16-31 hold K = k0+16..31
        const int kb = k0 + hl * 16;
        v16bf bh = *(const v16bf*)(wh + brow + kb);
        v16bf bl = *(const v16bf*)(wl + brow + kb);
        c = __builtin_amdgcn_wmma_f32_16x16x32_bf16(false, ah.v, false, bl, (short)0, c, false, false);
        c = __builtin_amdgcn_wmma_f32_16x16x32_bf16(false, al.v, false, bh, (short)0, c, false, false);
        c = __builtin_amdgcn_wmma_f32_16x16x32_bf16(false, ah.v, false, bh, (short)0, c, false, false);
    }
    // C/D: lane (l&15) = N, VGPR r = M (lanes<16: M=r, lanes>=16: M=r+8)
    const int n  = n0 + (lane & 15);
    const int mb = m0 + (hl ? 8 : 0);
#pragma unroll
    for (int r = 0; r < 8; ++r)
        out[(size_t)(mb + r) * D_DIM + n] = c[r];
}

// ---------------------------------------------------------------- sequential scan
// grid = (96 column-blocks, 4 batches), 256 threads = 8 waves, one column per wave.
// Each lane owns 24 rows (6 x float4, stride-128 chunks) of M and S in registers.
__global__ void __launch_bounds__(256)
scan_kernel(const float* __restrict__ qf, const float* __restrict__ kf,
            const float* __restrict__ vf, const float* __restrict__ gf,
            float* __restrict__ y, float* __restrict__ Mf, float* __restrict__ Sf) {
    extern __shared__ float smem[];
    float* k0b = smem;                  // [768]  double-buffered k/q staging
    float* q0b = k0b + D_DIM;
    float* k1b = q0b + D_DIM;
    float* q1b = k1b + D_DIM;

    const int tid  = threadIdx.x;
    const int lane = tid & 31;
    const int wave = tid >> 5;
    const int b    = blockIdx.y;
    const int col  = blockIdx.x * 8 + wave;

    float4 mreg[6], sreg[6];
#pragma unroll
    for (int i = 0; i < 6; ++i) {
        mreg[i] = make_float4(0.f, 0.f, 0.f, 0.f);
        sreg[i] = make_float4(0.f, 0.f, 0.f, 0.f);
    }

    // preload t = 0
    {
        size_t base = ((size_t)b * S_DIM) * D_DIM;
        for (int i = tid; i < D_DIM; i += 256) { k0b[i] = kf[base + i]; q0b[i] = qf[base + i]; }
    }
    size_t gb = (size_t)b * S_DIM;
    float v_cur = vf[gb * D_DIM + col];
    float a_cur = gf[gb * 3 + 0];
    float e_cur = gf[gb * 3 + 1];
    float t_cur = gf[gb * 3 + 2];
    __syncthreads();

    for (int t = 0; t < S_DIM; ++t) {
        float* kc = (t & 1) ? k1b : k0b;
        float* qc = (t & 1) ? q1b : q0b;
        float* kn = (t & 1) ? k0b : k1b;
        float* qn = (t & 1) ? q0b : q1b;

        // issue next step's staging + scalar prefetch first (overlaps with compute)
        const int tn = (t + 1 < S_DIM) ? t + 1 : t;
        const size_t basen = (gb + tn) * D_DIM;
        if (t + 1 < S_DIM) {
            for (int i = tid; i < D_DIM; i += 256) { kn[i] = kf[basen + i]; qn[i] = qf[basen + i]; }
        }
        float v_nxt = vf[basen + col];
        float a_nxt = gf[(gb + tn) * 3 + 0];
        float e_nxt = gf[(gb + tn) * 3 + 1];
        float t_nxt = gf[(gb + tn) * 3 + 2];

        // dot products against pre-update M (lane-local rows, then wave32 xor-reduce)
        float pk = 0.f, pq = 0.f;
        float4 kreg[6];
#pragma unroll
        for (int i = 0; i < 6; ++i) {
            const int r = 128 * i + 4 * lane;
            float4 k4 = *(const float4*)(kc + r);
            float4 q4 = *(const float4*)(qc + r);
            kreg[i] = k4;
            float4 m4 = mreg[i];
            pk += m4.x * k4.x + m4.y * k4.y + m4.z * k4.z + m4.w * k4.w;
            pq += m4.x * q4.x + m4.y * q4.y + m4.z * q4.z + m4.w * q4.w;
        }
#pragma unroll
        for (int off = 16; off >= 1; off >>= 1) {
            pk += __shfl_xor(pk, off, 32);
            pq += __shfl_xor(pq, off, 32);
        }

        const float err = pk - v_cur;
        if (lane == 0) y[(gb + t) * D_DIM + col] = pq;

        // S = eta*S - theta*(k (x) err);  M = (1-alpha)*M + S
        const float te = t_cur * err;
        const float cm = 1.0f - a_cur;
#pragma unroll
        for (int i = 0; i < 6; ++i) {
            float4 s4 = sreg[i], k4 = kreg[i], m4 = mreg[i];
            s4.x = e_cur * s4.x - te * k4.x;
            s4.y = e_cur * s4.y - te * k4.y;
            s4.z = e_cur * s4.z - te * k4.z;
            s4.w = e_cur * s4.w - te * k4.w;
            m4.x = cm * m4.x + s4.x;
            m4.y = cm * m4.y + s4.y;
            m4.z = cm * m4.z + s4.z;
            m4.w = cm * m4.w + s4.w;
            sreg[i] = s4; mreg[i] = m4;
        }
        v_cur = v_nxt; a_cur = a_nxt; e_cur = e_nxt; t_cur = t_nxt;
        __syncthreads();   // protects staging buffers (one barrier per step)
    }

    // write final M, S states
    float* Mp = Mf + (size_t)b * D_DIM * D_DIM;
    float* Sp = Sf + (size_t)b * D_DIM * D_DIM;
#pragma unroll
    for (int i = 0; i < 6; ++i) {
        const int r = 128 * i + 4 * lane;
        Mp[(size_t)(r + 0) * D_DIM + col] = mreg[i].x;
        Mp[(size_t)(r + 1) * D_DIM + col] = mreg[i].y;
        Mp[(size_t)(r + 2) * D_DIM + col] = mreg[i].z;
        Mp[(size_t)(r + 3) * D_DIM + col] = mreg[i].w;
        Sp[(size_t)(r + 0) * D_DIM + col] = sreg[i].x;
        Sp[(size_t)(r + 1) * D_DIM + col] = sreg[i].y;
        Sp[(size_t)(r + 2) * D_DIM + col] = sreg[i].z;
        Sp[(size_t)(r + 3) * D_DIM + col] = sreg[i].w;
    }
}

// ---------------------------------------------------------------- host launcher
extern "C" void kernel_launch(void* const* d_in, const int* in_sizes, int n_in,
                              void* d_out, int out_size, void* d_ws, size_t ws_size,
                              hipStream_t stream) {
    const float* x  = (const float*)d_in[0];
    const float* Wk = (const float*)d_in[1];
    const float* Wv = (const float*)d_in[2];
    const float* Wq = (const float*)d_in[3];
    const float* Wg = (const float*)d_in[4];
    const float* bg = (const float*)d_in[5];

    float* y  = (float*)d_out;
    float* Mf = y + (size_t)NTOK * D_DIM;
    float* Sf = Mf + (size_t)B_DIM * D_DIM * D_DIM;

    char* ws = (char*)d_ws;
    size_t off = 0;
    auto alloc = [&](size_t bytes) -> void* {
        void* p = ws + off;
        off += (bytes + 255) & ~(size_t)255;
        return p;
    };
    float* qf = (float*)alloc((size_t)NTOK * D_DIM * 4);
    float* kf = (float*)alloc((size_t)NTOK * D_DIM * 4);
    float* vf = (float*)alloc((size_t)NTOK * D_DIM * 4);
    float* gf = (float*)alloc((size_t)NTOK * 3 * 4);
    unsigned short* xh  = (unsigned short*)alloc((size_t)NTOK * D_DIM * 2);
    unsigned short* xl  = (unsigned short*)alloc((size_t)NTOK * D_DIM * 2);
    unsigned short* wkh = (unsigned short*)alloc((size_t)D_DIM * D_DIM * 2);
    unsigned short* wkl = (unsigned short*)alloc((size_t)D_DIM * D_DIM * 2);
    unsigned short* wvh = (unsigned short*)alloc((size_t)D_DIM * D_DIM * 2);
    unsigned short* wvl = (unsigned short*)alloc((size_t)D_DIM * D_DIM * 2);
    unsigned short* wqh = (unsigned short*)alloc((size_t)D_DIM * D_DIM * 2);
    unsigned short* wql = (unsigned short*)alloc((size_t)D_DIM * D_DIM * 2);

    // bf16 hi/lo splits
    {
        int n = NTOK * D_DIM;
        split_bf16_kernel<<<(n + 255) / 256, 256, 0, stream>>>(x, xh, xl, n);
        int m = D_DIM * D_DIM;
        split_bf16_kernel<<<(m + 255) / 256, 256, 0, stream>>>(Wk, wkh, wkl, m);
        split_bf16_kernel<<<(m + 255) / 256, 256, 0, stream>>>(Wv, wvh, wvl, m);
        split_bf16_kernel<<<(m + 255) / 256, 256, 0, stream>>>(Wq, wqh, wql, m);
    }
    // gates (full f32)
    gates_kernel<<<(NTOK * 3 + 255) / 256, 256, 0, stream>>>(x, Wg, bg, gf);

    // q/k/v projections: (8192/16)*(768/16) = 24576 tiles, 8 waves/block -> 3072 blocks
    const int gemm_blocks = (NTOK / 16) * (D_DIM / 16) / 8;
    proj_gemm_kernel<<<gemm_blocks, 256, 0, stream>>>(xh, xl, wkh, wkl, kf);
    proj_gemm_kernel<<<gemm_blocks, 256, 0, stream>>>(xh, xl, wvh, wvl, vf);
    proj_gemm_kernel<<<gemm_blocks, 256, 0, stream>>>(xh, xl, wqh, wql, qf);

    // scan: 96 column-blocks x 4 batches, 12 KB LDS (double-buffered k/q staging)
    dim3 sgrid(D_DIM / 8, B_DIM);
    size_t shmem = (size_t)4 * D_DIM * sizeof(float);
    scan_kernel<<<sgrid, 256, shmem, stream>>>(qf, kf, vf, gf, y, Mf, Sf);
}
